// LlamaAttention_41386304864888
// MI455X (gfx1250) — compile-verified, matrix operand order checked by reading, not run on hardware
//
#include <hip/hip_runtime.h>
#include <cstdint>
#include <cstddef>

// ---------------- types ----------------
typedef __bf16 bf16;
typedef __attribute__((ext_vector_type(16))) __bf16   v16bf;
typedef __attribute__((ext_vector_type(8)))  __bf16   v8bf;
typedef __attribute__((ext_vector_type(8)))  float    v8f;
typedef __attribute__((ext_vector_type(4)))  uint32_t u32x4;
typedef __attribute__((ext_vector_type(8)))  int      i32x8;
typedef __attribute__((ext_vector_type(4)))  int      i32x4;

#define WMMA_BF16(a, b, c) \
  __builtin_amdgcn_wmma_f32_16x16x32_bf16(false, (a), false, (b), (short)0, (c), false, false)

// Tensor Data Mover availability (6-arg form on this toolchain)
#if defined(__has_builtin)
#if __has_builtin(__builtin_amdgcn_tensor_load_to_lds) && \
    __has_builtin(__builtin_amdgcn_s_wait_tensorcnt)
#define USE_TDM 1
#endif
#endif
#ifndef USE_TDM
#define USE_TDM 0
#endif

// ---------------- problem constants ----------------
#define BB   2
#define SS   2048
#define HHID 2048
#define NH   32
#define NKV  8
#define HD   64
#define KVH  (NKV * HD)      // 512
#define MTOT (BB * SS)       // 4096

// ------------------------------------------------------------------
// Fragment loader for 16-bit A/B operands of v_wmma_*_16x16x32.
// lane L holds row (L&15); two contiguous 16B chunks at K offsets
// 8*(lane>>4) and +16.  B operands use the same pattern when B^T is
// row-major (K contiguous).  rowptr already points at this lane's row.
// ------------------------------------------------------------------
__device__ __forceinline__ v16bf load_frag(const bf16* rowptr, int k0) {
  const int half = (threadIdx.x >> 4) & 1;
  const bf16* p = rowptr + k0 + (half << 3);
  v8bf lo = *(const v8bf*)(p);
  v8bf hi = *(const v8bf*)(p + 16);
  v16bf r;
#pragma unroll
  for (int i = 0; i < 8; ++i) { r[i] = lo[i]; r[i + 8] = hi[i]; }
  return r;
}

// ---------------- fp32 -> bf16 convert ----------------
__global__ void f32_to_bf16_kernel(const float* __restrict__ in,
                                   bf16* __restrict__ out, int n) {
  int i = blockIdx.x * blockDim.x + threadIdx.x;
  int stride = gridDim.x * blockDim.x;
  for (; i < n; i += stride) out[i] = (bf16)in[i];
}

// ------------------------------------------------------------------
// C[M,N] (f32) = A[M,K] (bf16 row-major) @ B[N,K]^T (bf16 row-major)
// 128 threads = 4 waves; wave computes 32x64 (8 WMMA / k-step, 8 f32 accs),
// block computes 128x64.  A-fragments double-buffered in registers so loads
// overlap WMMA issue.  Working set (~150MB) fits the 192MB L2, so operands
// stream from global/L2 directly.
// ------------------------------------------------------------------
__global__ __launch_bounds__(128)
void gemm_bf16_wmma(const bf16* __restrict__ A, const bf16* __restrict__ B,
                    float* __restrict__ C, int M, int N, int K) {
  const int wave = threadIdx.x >> 5;
  const int lane = threadIdx.x & 31;
  const int half = lane >> 4;
  const int ln   = lane & 15;
  const int mbase = blockIdx.y * 128 + wave * 32;
  const int nbase = blockIdx.x * 64;

  const bf16* arow0 = A + (size_t)(mbase + ln) * K;
  const bf16* arow1 = A + (size_t)(mbase + 16 + ln) * K;
  const bf16* brow0 = B + (size_t)(nbase + 0 * 16 + ln) * K;
  const bf16* brow1 = B + (size_t)(nbase + 1 * 16 + ln) * K;
  const bf16* brow2 = B + (size_t)(nbase + 2 * 16 + ln) * K;
  const bf16* brow3 = B + (size_t)(nbase + 3 * 16 + ln) * K;

  v8f acc00 = {}, acc01 = {}, acc02 = {}, acc03 = {};
  v8f acc10 = {}, acc11 = {}, acc12 = {}, acc13 = {};

  v16bf a0 = load_frag(arow0, 0);
  v16bf a1 = load_frag(arow1, 0);

  for (int k0 = 0; k0 < K; k0 += 32) {
    if (k0 + 256 < K) {                 // gfx1250 global_prefetch_b8 path
      __builtin_prefetch(arow0 + k0 + 256, 0, 3);
      __builtin_prefetch(brow0 + k0 + 256, 0, 3);
    }
    v16bf b0 = load_frag(brow0, k0);
    v16bf b1 = load_frag(brow1, k0);
    v16bf b2 = load_frag(brow2, k0);
    v16bf b3 = load_frag(brow3, k0);

    v16bf na0 = a0, na1 = a1;
    if (k0 + 32 < K) {                  // register double-buffer of A
      na0 = load_frag(arow0, k0 + 32);
      na1 = load_frag(arow1, k0 + 32);
    }

    acc00 = WMMA_BF16(a0, b0, acc00);
    acc01 = WMMA_BF16(a0, b1, acc01);
    acc02 = WMMA_BF16(a0, b2, acc02);
    acc03 = WMMA_BF16(a0, b3, acc03);
    acc10 = WMMA_BF16(a1, b0, acc10);
    acc11 = WMMA_BF16(a1, b1, acc11);
    acc12 = WMMA_BF16(a1, b2, acc12);
    acc13 = WMMA_BF16(a1, b3, acc13);

    a0 = na0;
    a1 = na1;
  }

  // C/D layout: element (m,n): lane = (n&15) + 16*(m>=8), vgpr = m&7
#pragma unroll
  for (int r = 0; r < 8; ++r) {
    float* crow0 = C + (size_t)(mbase + r + half * 8) * N + nbase;
    crow0[0 * 16 + ln] = acc00[r];
    crow0[1 * 16 + ln] = acc01[r];
    crow0[2 * 16 + ln] = acc02[r];
    crow0[3 * 16 + ln] = acc03[r];
    float* crow1 = C + (size_t)(mbase + 16 + r + half * 8) * N + nbase;
    crow1[0 * 16 + ln] = acc10[r];
    crow1[1 * 16 + ln] = acc11[r];
    crow1[2 * 16 + ln] = acc12[r];
    crow1[3 * 16 + ln] = acc13[r];
  }
}

// ------------------------------------------------------------------
// RoPE + head-permute: projection outputs (f32) -> bf16 [b, head, s, d].
// ------------------------------------------------------------------
__global__ void rope_permute_kernel(const float* __restrict__ Qf,
                                    const float* __restrict__ Kf,
                                    const float* __restrict__ Vf,
                                    const int*   __restrict__ pos,
                                    bf16* __restrict__ qb,
                                    bf16* __restrict__ kb,
                                    bf16* __restrict__ vb) {
  const int total = BB * SS * (NH + 2 * NKV) * HD;   // hh in [0,48)
  for (int idx = blockIdx.x * blockDim.x + threadIdx.x; idx < total;
       idx += gridDim.x * blockDim.x) {
    const int d  = idx & (HD - 1);
    const int hh = (idx >> 6) % 48;
    const int s  = (idx >> 6) / 48 % SS;
    const int b  = idx / (48 * HD * SS);

    const int   i   = d & 31;
    const float ang = (float)pos[s] * __powf(10000.0f, -(float)i * (1.0f / 32.0f));
    const float c   = __cosf(ang);
    const float sn  = __sinf(ang);
    const float sgn = (d < 32) ? -1.0f : 1.0f;

    if (hh < NH) {                                   // Q
      const float* src = Qf + (size_t)(b * SS + s) * HHID + hh * HD;
      float x  = src[d];
      float xp = src[d ^ 32];
      qb[((size_t)(b * NH + hh) * SS + s) * HD + d] = (bf16)(x * c + sgn * xp * sn);
    } else if (hh < NH + NKV) {                      // K
      const int kh = hh - NH;
      const float* src = Kf + (size_t)(b * SS + s) * KVH + kh * HD;
      float x  = src[d];
      float xp = src[d ^ 32];
      kb[((size_t)(b * NKV + kh) * SS + s) * HD + d] = (bf16)(x * c + sgn * xp * sn);
    } else {                                         // V (convert only)
      const int vh = hh - NH - NKV;
      vb[((size_t)(b * NKV + vh) * SS + s) * HD + d] =
          (bf16)Vf[(size_t)(b * SS + s) * KVH + vh * HD + d];
    }
  }
}

// ------------------------------------------------------------------
// Flash attention (causal, GQA 4:1).  1 WG = 4 waves = 64 query rows of one
// (b,h).  K tile (32 keys x 64 dims, 4KB) is fetched by the Tensor Data
// Mover (tensor_load_to_lds + s_wait_tensorcnt) when available; V^T is
// staged manually (TDM has no transpose).
// ------------------------------------------------------------------
__global__ __launch_bounds__(128)
void flash_attn_kernel(const bf16* __restrict__ q, const bf16* __restrict__ k,
                       const bf16* __restrict__ v, bf16* __restrict__ ctx) {
  __shared__ __align__(16) bf16 Kt[32][HD];      // [key][dim]
  __shared__ __align__(16) bf16 Vt[HD][32];      // [dim][key]  (V^T)
  __shared__ __align__(16) bf16 Pt[4][16][32];   // per-wave P staging

  const int qblk = blockIdx.x % (SS / 64);
  const int h    = (blockIdx.x / (SS / 64)) % NH;
  const int b    = blockIdx.x / ((SS / 64) * NH);
  const int kvh  = h >> 2;
  const int wave = threadIdx.x >> 5;
  const int lane = threadIdx.x & 31;
  const int half = lane >> 4;
  const int ln   = lane & 15;
  const int qb   = qblk * 64;

  const bf16* qrow = q + ((size_t)(b * NH + h) * SS + qb + wave * 16 + ln) * HD;
  const v16bf aQ0 = load_frag(qrow, 0);
  const v16bf aQ1 = load_frag(qrow, 32);

  const bf16* kbp = k + (size_t)(b * NKV + kvh) * SS * HD;
  const bf16* vbp = v + (size_t)(b * NKV + kvh) * SS * HD;

  v8f o0 = {}, o1 = {}, o2 = {}, o3 = {};
  float mrow[8], lrow[8];
#pragma unroll
  for (int r = 0; r < 8; ++r) { mrow[r] = -1e30f; lrow[r] = 0.0f; }

  const int ntiles = (qb + 64) / 32;               // causal bound (uniform)

  for (int t = 0; t < ntiles; ++t) {
    const int kb0 = t * 32;

#if USE_TDM
    if (wave == 0) {
      // ---- 1-D TDM descriptor: 2048 bf16 elements global -> LDS ----
      const uint64_t ga = (uint64_t)(uintptr_t)(kbp + (size_t)kb0 * HD);
      const uint32_t la = (uint32_t)(uintptr_t)(&Kt[0][0]);   // LDS byte offset
      u32x4 g0;
      g0[0] = 1u;                                  // count=1 valid descriptor
      g0[1] = la;                                  // lds_addr
      g0[2] = (uint32_t)ga;                        // global_addr[31:0]
      g0[3] = (uint32_t)(ga >> 32) | (2u << 30);   // global_addr[56:32] | type=2
      i32x8 g1;
      g1[0] = 0x00010000;                          // data_size=2B; no flags
      g1[1] = 0x08000000;                          // tensor_dim0=2048 (bits 63:48)
      g1[2] = 0x00010000;                          // tensor_dim1=1   (bits 95:80)
      g1[3] = 0x08000000;                          // tile_dim0=2048  (bits 127:112)
      g1[4] = 0;                                   // tile_dim1=0, tile_dim2=0
      g1[5] = 2048;                                // tensor_dim0_stride lo
      g1[6] = 0;
      g1[7] = 0;
      i32x4 gz4 = {0, 0, 0, 0};
      i32x8 gz8 = {0, 0, 0, 0, 0, 0, 0, 0};
      __builtin_amdgcn_tensor_load_to_lds(g0, g1, gz4, gz4, gz8, 0);
      __builtin_amdgcn_s_wait_tensorcnt(0);
    }
    {  // cooperative V^T tile load (manual: needs transpose)
      const int row = threadIdx.x >> 2;
      const int cch = threadIdx.x & 3;
      const bf16* vs = vbp + (size_t)(kb0 + row) * HD + cch * 16;
#pragma unroll
      for (int e = 0; e < 16; ++e) Vt[cch * 16 + e][row] = vs[e];
    }
#else
    {  // cooperative K tile + V^T tile load (fallback path)
      const int row = threadIdx.x >> 2;            // 0..31 (key)
      const int cch = threadIdx.x & 3;             // 16-dim chunk
      const bf16* ks = kbp + (size_t)(kb0 + row) * HD + cch * 16;
      *(u32x4*)&Kt[row][cch * 16]     = *(const u32x4*)(ks);
      *(u32x4*)&Kt[row][cch * 16 + 8] = *(const u32x4*)(ks + 8);
      const bf16* vs = vbp + (size_t)(kb0 + row) * HD + cch * 16;
#pragma unroll
      for (int e = 0; e < 16; ++e) Vt[cch * 16 + e][row] = vs[e];
    }
#endif
    __syncthreads();

    // ---- scores: S = Q @ K^T  (16q x 32k as two 16x16 f32 tiles) ----
    v8f s0 = {}, s1 = {};
    {
      v16bf bk = load_frag(&Kt[ln][0], 0);       s0 = WMMA_BF16(aQ0, bk, s0);
      bk       = load_frag(&Kt[ln][0], 32);      s0 = WMMA_BF16(aQ1, bk, s0);
      bk       = load_frag(&Kt[16 + ln][0], 0);  s1 = WMMA_BF16(aQ0, bk, s1);
      bk       = load_frag(&Kt[16 + ln][0], 32); s1 = WMMA_BF16(aQ1, bk, s1);
    }

    // ---- online softmax (rows live in 16-lane halves; xor 1/2/4/8) ----
#pragma unroll
    for (int r = 0; r < 8; ++r) {
      const int qi = qb + wave * 16 + r + half * 8;
      float sc0 = s0[r] * 0.125f;
      float sc1 = s1[r] * 0.125f;
      if (kb0 + ln > qi)      sc0 = -1e30f;        // causal mask (lane select)
      if (kb0 + 16 + ln > qi) sc1 = -1e30f;

      float tmax = fmaxf(sc0, sc1);
#pragma unroll
      for (int off = 1; off < 16; off <<= 1)
        tmax = fmaxf(tmax, __shfl_xor(tmax, off));

      const float mnew  = fmaxf(mrow[r], tmax);
      const float alpha = __expf(mrow[r] - mnew);
      const float p0 = __expf(sc0 - mnew);
      const float p1 = __expf(sc1 - mnew);

      float psum = p0 + p1;
#pragma unroll
      for (int off = 1; off < 16; off <<= 1)
        psum += __shfl_xor(psum, off);

      lrow[r] = lrow[r] * alpha + psum;
      mrow[r] = mnew;
      o0[r] *= alpha; o1[r] *= alpha; o2[r] *= alpha; o3[r] *= alpha;

      // stage P (C-layout -> LDS row-major) for A-fragment reload
      Pt[wave][r + half * 8][ln]      = (bf16)p0;
      Pt[wave][r + half * 8][16 + ln] = (bf16)p1;
    }

    // ---- O += P @ V  (K = 32 keys, 4 dim tiles) ----
    {
      const v16bf aP = load_frag(&Pt[wave][ln][0], 0);
      v16bf bv = load_frag(&Vt[ln][0], 0);      o0 = WMMA_BF16(aP, bv, o0);
      bv       = load_frag(&Vt[16 + ln][0], 0); o1 = WMMA_BF16(aP, bv, o1);
      bv       = load_frag(&Vt[32 + ln][0], 0); o2 = WMMA_BF16(aP, bv, o2);
      bv       = load_frag(&Vt[48 + ln][0], 0); o3 = WMMA_BF16(aP, bv, o3);
    }
    __syncthreads();
  }

  // ---- normalize + store context as bf16 [b, s, h*64+d] ----
#pragma unroll
  for (int r = 0; r < 8; ++r) {
    const int qi = qb + wave * 16 + r + half * 8;
    const float inv = 1.0f / lrow[r];
    bf16* dst = ctx + ((size_t)(b * SS) + qi) * HHID + h * HD;
    dst[0 * 16 + ln] = (bf16)(o0[r] * inv);
    dst[1 * 16 + ln] = (bf16)(o1[r] * inv);
    dst[2 * 16 + ln] = (bf16)(o2[r] * inv);
    dst[3 * 16 + ln] = (bf16)(o3[r] * inv);
  }
}

// ------------------------------------------------------------------
extern "C" void kernel_launch(void* const* d_in, const int* in_sizes, int n_in,
                              void* d_out, int out_size, void* d_ws, size_t ws_size,
                              hipStream_t stream) {
  (void)in_sizes; (void)n_in; (void)out_size; (void)ws_size;
  const float* X   = (const float*)d_in[0];
  const float* Wq  = (const float*)d_in[1];
  const float* Wk  = (const float*)d_in[2];
  const float* Wv  = (const float*)d_in[3];
  const float* Wo  = (const float*)d_in[4];
  const int*   pos = (const int*)d_in[5];
  // d_in[6] = attention_mask: pure causal, implemented analytically.

  char* ws = (char*)d_ws;
  size_t off = 0;
  auto alloc = [&](size_t bytes) -> void* {
    void* p = ws + off;
    off = (off + bytes + 255) & ~(size_t)255;
    return p;
  };

  bf16*  Xb  = (bf16*) alloc((size_t)MTOT * HHID * 2);   // aliased later as ctx
  bf16*  Wqb = (bf16*) alloc((size_t)HHID * HHID * 2);
  bf16*  Wkb = (bf16*) alloc((size_t)KVH  * HHID * 2);
  bf16*  Wvb = (bf16*) alloc((size_t)KVH  * HHID * 2);
  bf16*  Wob = (bf16*) alloc((size_t)HHID * HHID * 2);
  float* Qf  = (float*)alloc((size_t)MTOT * HHID * 4);
  float* Kf  = (float*)alloc((size_t)MTOT * KVH  * 4);
  float* Vf  = (float*)alloc((size_t)MTOT * KVH  * 4);
  bf16*  qbf = (bf16*) alloc((size_t)MTOT * HHID * 2);   // [B,NH,S,D]
  bf16*  kbf = (bf16*) alloc((size_t)MTOT * KVH  * 2);   // [B,NKV,S,D]
  bf16*  vbf = (bf16*) alloc((size_t)MTOT * KVH  * 2);   // [B,NKV,S,D]
  bf16*  ctx = Xb;  // X-bf16 is dead once projections complete

  // 1) fp32 -> bf16 converts
  f32_to_bf16_kernel<<<2048, 256, 0, stream>>>(X,  Xb,  MTOT * HHID);
  f32_to_bf16_kernel<<<2048, 256, 0, stream>>>(Wq, Wqb, HHID * HHID);
  f32_to_bf16_kernel<<<1024, 256, 0, stream>>>(Wk, Wkb, KVH * HHID);
  f32_to_bf16_kernel<<<1024, 256, 0, stream>>>(Wv, Wvb, KVH * HHID);
  f32_to_bf16_kernel<<<2048, 256, 0, stream>>>(Wo, Wob, HHID * HHID);

  // 2) projections:  P = X @ W^T   (A = X[M,K], B = W[N,K])
  gemm_bf16_wmma<<<dim3(HHID / 64, MTOT / 128), 128, 0, stream>>>(Xb, Wqb, Qf, MTOT, HHID, HHID);
  gemm_bf16_wmma<<<dim3(KVH  / 64, MTOT / 128), 128, 0, stream>>>(Xb, Wkb, Kf, MTOT, KVH,  HHID);
  gemm_bf16_wmma<<<dim3(KVH  / 64, MTOT / 128), 128, 0, stream>>>(Xb, Wvb, Vf, MTOT, KVH,  HHID);

  // 3) RoPE + permute to [b, head, s, d] bf16
  rope_permute_kernel<<<2048, 256, 0, stream>>>(Qf, Kf, Vf, pos, qbf, kbf, vbf);

  // 4) causal flash attention -> ctx bf16 [b, s, h*d]
  flash_attn_kernel<<<BB * NH * (SS / 64), 128, 0, stream>>>(qbf, kbf, vbf, ctx);

  // 5) output projection -> d_out fp32 [B, S, H]
  gemm_bf16_wmma<<<dim3(HHID / 64, MTOT / 128), 128, 0, stream>>>(ctx, Wob, (float*)d_out,
                                                                  MTOT, HHID, HHID);
}
